// GCNIED_88278757802608
// MI455X (gfx1250) — matrix operand order repaired and checked by reading, not compiled, not run on hardware
//
#include <hip/hip_runtime.h>
#include <hip/hip_bf16.h>
#include <math.h>

// ---------------- problem constants (from reference) ----------------
#define N_NODES 8192
#define NFEAT   512
#define NHID    256
#define NCLASS  40
#define NNZ     131072
#define HOPS    4
#define KTOP    8
#define THRESH  0.6f
#define LAMBD   (1.0f / (8.0f + 1.0f / (0.6f - 0.3f)))   // 0.0882353
#define NE_MAX  (NNZ + N_NODES * KTOP)                    // union pattern upper bound

typedef __bf16 bf16_t;
typedef __attribute__((ext_vector_type(16))) __bf16 v16bf;
typedef __attribute__((ext_vector_type(8)))  float  v8f;

// round-to-nearest-even f32 -> bf16 (pure integer, no __bf16 arithmetic needed)
__device__ __forceinline__ bf16_t f2bf(float f) {
  unsigned u = __float_as_uint(f);
  unsigned r = u + 0x7FFFu + ((u >> 16) & 1u);
  unsigned short h = (unsigned short)(r >> 16);
  bf16_t o;
  __builtin_memcpy(&o, &h, 2);
  return o;
}

// ---------------- WMMA fragment loaders (contiguous k order, same for A and B) ----------------
// A (16x32 bf16): lane L holds row m=L&15; halfs i -> K = 16*(L>=16) + i
__device__ __forceinline__ v16bf load_frag_a(const bf16_t* __restrict__ rowp, int lane) {
  const int g = (lane >> 4) << 4;
  v16bf a;
#pragma unroll
  for (int i = 0; i < 16; ++i) a[i] = rowp[g + i];
  return a;
}
// B (32x16 bf16) loaded from B-transposed [n][k]: lane L holds col n=L&15;
// halfs i -> K = 16*(L>=16) + i  (16 contiguous halfs -> one 32B load)
__device__ __forceinline__ v16bf load_frag_b(const bf16_t* __restrict__ rowp, int lane) {
  const int g = (lane >> 4) << 4;
  v16bf b;
#pragma unroll
  for (int i = 0; i < 16; ++i) b[i] = rowp[g + i];
  return b;
}

// ---------------- generic bf16 WMMA GEMM: C[M,N] = act(A[M,K] @ Bt[N,K]^T + bias) ----------------
// block = 256 threads = 8 waves; wave w -> 16x(16*NT) tile: A fragment reused across NT wmma's
template <int NT>
__global__ __launch_bounds__(256) void gemm_bf16_wmma(
    const bf16_t* __restrict__ A, const bf16_t* __restrict__ Bt,
    const float* __restrict__ bias, float* __restrict__ C,
    int K, int Nvalid, int ldc, int relu) {
  const int lane = threadIdx.x & 31;
  const int wave = threadIdx.x >> 5;
  const int m0 = (blockIdx.y * 8 + wave) * 16;
  const int n0 = blockIdx.x * (16 * NT);
  const bf16_t* arow = A + (size_t)(m0 + (lane & 15)) * K;
  const bf16_t* brow[NT];
#pragma unroll
  for (int nt = 0; nt < NT; ++nt)
    brow[nt] = Bt + (size_t)(n0 + nt * 16 + (lane & 15)) * K;
  const v8f zero = {0.f, 0.f, 0.f, 0.f, 0.f, 0.f, 0.f, 0.f};
  v8f c[NT];
#pragma unroll
  for (int nt = 0; nt < NT; ++nt) c[nt] = zero;
  for (int k0 = 0; k0 < K; k0 += 32) {
    v16bf a = load_frag_a(arow + k0, lane);
#pragma unroll
    for (int nt = 0; nt < NT; ++nt) {
      v16bf b = load_frag_b(brow[nt] + k0, lane);
      c[nt] = __builtin_amdgcn_wmma_f32_16x16x32_bf16(false, a, false, b, (short)0, c[nt], false, false);
    }
  }
  const int mb = (lane >> 4) << 3;
#pragma unroll
  for (int nt = 0; nt < NT; ++nt) {
    const int n = n0 + nt * 16 + (lane & 15);
    if (n >= Nvalid) continue;
    const float bv = bias ? bias[n] : 0.0f;
#pragma unroll
    for (int j = 0; j < 8; ++j) {
      float v = c[nt][j] + bv;
      if (relu) v = fmaxf(v, 0.0f);
      C[(size_t)(m0 + mb + j) * ldc + n] = v;
    }
  }
}

// ---------------- conversion / transpose helpers ----------------
__global__ void f32_to_bf16_kernel(const float* __restrict__ src, bf16_t* __restrict__ dst, int n) {
  int i = blockIdx.x * blockDim.x + threadIdx.x;
  if (i < n) dst[i] = f2bf(src[i]);
}
// src[rows][cols] f32 -> dst[colsPad][rows] bf16 (zero-padded columns)
__global__ void transpose_pad_bf16_kernel(const float* __restrict__ src, bf16_t* __restrict__ dst,
                                          int rows, int cols, int colsPad) {
  int idx = blockIdx.x * blockDim.x + threadIdx.x;
  if (idx >= colsPad * rows) return;
  int n = idx / rows, k = idx % rows;
  float v = (n < cols) ? src[(size_t)k * cols + n] : 0.0f;
  dst[idx] = f2bf(v);
}

// ---------------- sparse adj SPMM: Y[row] += val * X[col]  (wave per edge) ----------------
__global__ void spmm_edge_kernel(const int* __restrict__ rows, const int* __restrict__ cols,
                                 const float* __restrict__ vals,
                                 const float* __restrict__ X, float* __restrict__ Y) {
  int e = (int)((blockIdx.x * blockDim.x + threadIdx.x) >> 5);
  int lane = threadIdx.x & 31;
  if (e >= NNZ) return;
  int r = rows[e], c = cols[e];
  float v = vals[e];
  const float* xr = X + (size_t)c * NHID + lane * 8;
  float* yr = Y + (size_t)r * NHID + lane * 8;
#pragma unroll
  for (int i = 0; i < 8; ++i) atomicAdd(&yr[i], v * xr[i]);
}

// ---------------- row L2-normalize: Xn (f32) + Xnb (bf16) ----------------
__global__ void rownorm_kernel(const float* __restrict__ X, float* __restrict__ Xn,
                               bf16_t* __restrict__ Xnb) {
  __shared__ float red[256];
  int row = blockIdx.x, d = threadIdx.x;
  float v = X[(size_t)row * NHID + d];
  red[d] = v * v;
  __syncthreads();
  for (int s = 128; s > 0; s >>= 1) { if (d < s) red[d] += red[d + s]; __syncthreads(); }
  float nrm = fmaxf(sqrtf(red[0]), 1e-12f);
  float o = v / nrm;
  Xn[(size_t)row * NHID + d] = o;
  Xnb[(size_t)row * NHID + d] = f2bf(o);
}

// ---------------- fused cos = Xn@Xn^T (WMMA) + streaming per-row top-8 ----------------
// 32 rows per block (2 m-tiles per wave): halves L2 B-panel traffic vs 16-row blocks
// and amortizes each B fragment over 2 WMMAs.
__device__ __forceinline__ void top8_insert(float (&tv)[8], int (&ti)[8], float v, int j) {
  if (v <= tv[7]) return;
  tv[7] = v; ti[7] = j;
#pragma unroll
  for (int s = 7; s >= 1; --s) {
    if (tv[s] > tv[s - 1]) {
      float a = tv[s]; tv[s] = tv[s - 1]; tv[s - 1] = a;
      int b = ti[s]; ti[s] = ti[s - 1]; ti[s - 1] = b;
    }
  }
}

__global__ __launch_bounds__(256) void cos_topk_kernel(const bf16_t* __restrict__ Xnb,
                                                       float* __restrict__ tkv,
                                                       int* __restrict__ tki) {
  __shared__ float tile[8][2][16][17];  // per-wave, per m-tile 16x16 cos tile (padded)
  __shared__ float cv[32][128];         // merge candidates: 16 lanes/row * top-8
  __shared__ int   ci[32][128];
  const int lane = threadIdx.x & 31;
  const int wave = threadIdx.x >> 5;
  const int r0 = blockIdx.x * 32;
  const bf16_t* arow0 = Xnb + (size_t)(r0 + (lane & 15)) * NHID;
  const bf16_t* arow1 = arow0 + (size_t)16 * NHID;
  const int myrow = lane & 15;          // row within an m-tile
  const int cb = (lane >> 4) << 3;      // this lane scans cols cb..cb+7 of each tile
  float tv[2][8]; int ti[2][8];
#pragma unroll
  for (int mt = 0; mt < 2; ++mt)
#pragma unroll
    for (int s = 0; s < 8; ++s) { tv[mt][s] = -3.0e38f; ti[mt][s] = 0; }

  for (int t = wave; t < N_NODES / 16; t += 8) {     // 64 uniform iterations per wave
    const bf16_t* brow = Xnb + (size_t)(t * 16 + (lane & 15)) * NHID;
    v8f c0 = {0.f, 0.f, 0.f, 0.f, 0.f, 0.f, 0.f, 0.f};
    v8f c1 = c0;
#pragma unroll
    for (int kk = 0; kk < NHID; kk += 32) {
      v16bf b  = load_frag_b(brow + kk, lane);       // shared across both m-tiles
      v16bf a0 = load_frag_a(arow0 + kk, lane);
      v16bf a1 = load_frag_a(arow1 + kk, lane);
      c0 = __builtin_amdgcn_wmma_f32_16x16x32_bf16(false, a0, false, b, (short)0, c0, false, false);
      c1 = __builtin_amdgcn_wmma_f32_16x16x32_bf16(false, a1, false, b, (short)0, c1, false, false);
    }
    { // spill D tiles (lane owns col n, rows mb..mb+7) into wave-private LDS
      const int n = lane & 15, mb = (lane >> 4) << 3;
#pragma unroll
      for (int j = 0; j < 8; ++j) tile[wave][0][mb + j][n] = c0[j];
#pragma unroll
      for (int j = 0; j < 8; ++j) tile[wave][1][mb + j][n] = c1[j];
    }
    __syncthreads();
#pragma unroll
    for (int mt = 0; mt < 2; ++mt) {
#pragma unroll
      for (int q = 0; q < 8; ++q) {
        int col = t * 16 + cb + q;
        float v = tile[wave][mt][myrow][cb + q];
        if (r0 + mt * 16 + myrow == col) v -= 1.0f;  // newcos = cos - eye
        top8_insert(tv[mt], ti[mt], v, col);
      }
    }
    __syncthreads();
  }
  // dump per-lane top-8 candidates, then per-row argmax selection of final 8
  const int slot = (wave * 2 + (lane >> 4)) * 8;
#pragma unroll
  for (int mt = 0; mt < 2; ++mt)
#pragma unroll
    for (int s = 0; s < 8; ++s) {
      cv[mt * 16 + myrow][slot + s] = tv[mt][s];
      ci[mt * 16 + myrow][slot + s] = ti[mt][s];
    }
  __syncthreads();
  if (threadIdx.x < 32) {
    int row = threadIdx.x;
    for (int kk = 0; kk < KTOP; ++kk) {
      float best = -3.0e38f; int bs = 0;
      for (int s = 0; s < 128; ++s) { float x = cv[row][s]; if (x > best) { best = x; bs = s; } }
      tkv[(r0 + row) * KTOP + kk] = best;
      tki[(r0 + row) * KTOP + kk] = ci[row][bs];
      cv[row][bs] = -3.0e38f;
    }
  }
}

// ---------------- per-edge cosine + threshold (wave per edge) ----------------
__global__ void edge_val_kernel(const int* __restrict__ rows, const int* __restrict__ cols,
                                const float* __restrict__ vals,
                                const float* __restrict__ Xn, float* __restrict__ ea) {
  int e = (int)((blockIdx.x * blockDim.x + threadIdx.x) >> 5);
  int lane = threadIdx.x & 31;
  if (e >= NNZ) return;
  const float* xr = Xn + (size_t)rows[e] * NHID + lane * 8;
  const float* xc = Xn + (size_t)cols[e] * NHID + lane * 8;
  float acc = 0.f;
#pragma unroll
  for (int i = 0; i < 8; ++i) acc += xr[i] * xc[i];
#pragma unroll
  for (int m = 16; m >= 1; m >>= 1) acc += __shfl_xor(acc, m, 32);
  if (lane == 0) ea[e] = (acc < THRESH) ? -1.0e9f : vals[e];
}

// ---------------- CSR build for union pattern ----------------
__global__ void count_edges_kernel(const int* __restrict__ rows, int* __restrict__ cnt) {
  int e = blockIdx.x * blockDim.x + threadIdx.x;
  if (e < NNZ) atomicAdd(&cnt[rows[e]], 1);
}
__global__ void count_topk_kernel(const float* __restrict__ tkv, int* __restrict__ cnt) {
  int i = blockIdx.x * blockDim.x + threadIdx.x;
  if (i < N_NODES * KTOP && tkv[i] != 0.0f) atomicAdd(&cnt[i / KTOP], 1);
}
__global__ __launch_bounds__(1024) void scan_kernel(const int* __restrict__ cnt,
                                                    int* __restrict__ rp, int* __restrict__ cur) {
  __shared__ int s[1024];
  int t = threadIdx.x;
  int base = t * 8, sum = 0;
#pragma unroll
  for (int i = 0; i < 8; ++i) sum += cnt[base + i];
  s[t] = sum;
  __syncthreads();
  for (int off = 1; off < 1024; off <<= 1) {
    int v = (t >= off) ? s[t - off] : 0;
    __syncthreads();
    s[t] += v;
    __syncthreads();
  }
  int run = (t == 0) ? 0 : s[t - 1];
  for (int i = 0; i < 8; ++i) {
    rp[base + i] = run;
    cur[base + i] = run;
    run += cnt[base + i];
  }
  if (t == 1023) rp[N_NODES] = run;
}
__global__ void fill_edges_kernel(const int* __restrict__ rows, const int* __restrict__ cols,
                                  const float* __restrict__ eaIn, int* __restrict__ cur,
                                  int* __restrict__ ecol, float* __restrict__ ea) {
  int e = blockIdx.x * blockDim.x + threadIdx.x;
  if (e >= NNZ) return;
  int p = atomicAdd(&cur[rows[e]], 1);
  ecol[p] = cols[e];
  ea[p] = eaIn[e];
}
__global__ void fill_topk_kernel(const float* __restrict__ tkv, const int* __restrict__ tki,
                                 int* __restrict__ cur, int* __restrict__ ecol,
                                 float* __restrict__ ea) {
  int i = blockIdx.x * blockDim.x + threadIdx.x;
  if (i >= N_NODES * KTOP) return;
  float v = tkv[i];
  if (v == 0.0f) return;                 // to_sparse() drops exact zeros
  int p = atomicAdd(&cur[i / KTOP], 1);
  ecol[p] = tki[i];
  ea[p] = LAMBD * v;
}

// ---------------- sparse per-row softmax (wave per row) ----------------
__global__ void row_softmax_kernel(const int* __restrict__ rp, const float* __restrict__ ea,
                                   float* __restrict__ ew) {
  int row = (int)((blockIdx.x * blockDim.x + threadIdx.x) >> 5);
  int lane = threadIdx.x & 31;
  if (row >= N_NODES) return;
  int s0 = rp[row], s1 = rp[row + 1];
  if (s1 == s0) return;
  float mx = -3.0e38f;
  for (int p = s0 + lane; p < s1; p += 32) mx = fmaxf(mx, ea[p]);
#pragma unroll
  for (int m = 16; m >= 1; m >>= 1) mx = fmaxf(mx, __shfl_xor(mx, m, 32));
  float sum = 0.f;
  for (int p = s0 + lane; p < s1; p += 32) sum += __expf(ea[p] - mx);
#pragma unroll
  for (int m = 16; m >= 1; m >>= 1) sum += __shfl_xor(sum, m, 32);
  float inv = 1.0f / fmaxf(sum, 1e-16f);
  for (int p = s0 + lane; p < s1; p += 32) ew[p] = __expf(ea[p] - mx) * inv;
}

// ---------------- sparse propagation hop: Y[row] = sum_e w_e * X[col_e] ----------------
__global__ void prop_kernel(const int* __restrict__ rp, const int* __restrict__ ecol,
                            const float* __restrict__ ew,
                            const float* __restrict__ X, float* __restrict__ Y) {
  int row = blockIdx.x, d = threadIdx.x;
  int s0 = rp[row], s1 = rp[row + 1];
  float acc = 0.f;
  for (int p = s0; p < s1; ++p) {
    if (p + 1 < s1) __builtin_prefetch(&X[(size_t)ecol[p + 1] * NHID + d], 0, 1);
    acc += ew[p] * X[(size_t)ecol[p] * NHID + d];
  }
  Y[(size_t)row * NHID + d] = acc;
}

// ---------------- per-node hop attention + combine -> bf16 ----------------
__global__ void hop_attn_kernel(const float* __restrict__ f0, const float* __restrict__ f1,
                                const float* __restrict__ f2, const float* __restrict__ f3,
                                const float* __restrict__ f4, const float* __restrict__ propW,
                                bf16_t* __restrict__ combbf) {
  __shared__ float red[256];
  __shared__ float coef[5];
  int row = blockIdx.x, d = threadIdx.x;
  size_t off = (size_t)row * NHID + d;
  float fv[5] = { f0[off], f1[off], f2[off], f3[off], f4[off] };
  float pv = propW[off];
  float sc[5];
  for (int h = 0; h < 5; ++h) {
    red[d] = fv[h] * pv;
    __syncthreads();
    for (int s = 128; s > 0; s >>= 1) { if (d < s) red[d] += red[d + s]; __syncthreads(); }
    sc[h] = red[0];
    __syncthreads();
  }
  if (d == 0) {
    float mx = sc[0];
    for (int h = 1; h < 5; ++h) mx = fmaxf(mx, sc[h]);
    float sm = 0.f;
    for (int h = 0; h < 5; ++h) { coef[h] = __expf(sc[h] - mx); sm += coef[h]; }
    for (int h = 0; h < 5; ++h) coef[h] /= sm;
  }
  __syncthreads();
  float o = 0.f;
  for (int h = 0; h < 5; ++h) o += coef[h] * fv[h];
  combbf[off] = f2bf(o);
}

// =========================== host orchestration ===========================
extern "C" void kernel_launch(void* const* d_in, const int* in_sizes, int n_in,
                              void* d_out, int out_size, void* d_ws, size_t ws_size,
                              hipStream_t stream) {
  (void)in_sizes; (void)n_in; (void)out_size; (void)ws_size;
  const float* x    = (const float*)d_in[0];
  const int*   adj  = (const int*)d_in[1];
  const float* avals= (const float*)d_in[2];
  const float* W1   = (const float*)d_in[3];
  const float* b1   = (const float*)d_in[4];
  const float* W2   = (const float*)d_in[5];
  const float* b2   = (const float*)d_in[6];
  const float* prp  = (const float*)d_in[7];
  const int* erow = adj;
  const int* ecolIn = adj + NNZ;
  float* out = (float*)d_out;

  char* w = (char*)d_ws;
  auto alloc = [&](size_t bytes) -> void* {
    void* p = (void*)w; w += (bytes + 255) & ~(size_t)255; return p;
  };
  bf16_t* x_bf   = (bf16_t*)alloc((size_t)N_NODES * NFEAT * 2);
  bf16_t* W1t_bf = (bf16_t*)alloc((size_t)NHID * NFEAT * 2);      // [256][512]
  bf16_t* W2t_bf = (bf16_t*)alloc((size_t)48 * NHID * 2);         // [48][256] zero-padded
  float*  h      = (float*)alloc((size_t)N_NODES * NHID * 4);     // feats[0]
  float*  xa     = (float*)alloc((size_t)N_NODES * NHID * 4);
  float*  xb     = (float*)alloc((size_t)N_NODES * NHID * 4);
  float*  xn     = (float*)alloc((size_t)N_NODES * NHID * 4);
  bf16_t* xn_bf  = (bf16_t*)alloc((size_t)N_NODES * NHID * 2);
  float*  tkv    = (float*)alloc((size_t)N_NODES * KTOP * 4);
  int*    tki    = (int*)alloc((size_t)N_NODES * KTOP * 4);
  float*  eedge  = (float*)alloc((size_t)NNZ * 4);
  int*    cnt    = (int*)alloc((size_t)N_NODES * 4);
  int*    cur    = (int*)alloc((size_t)N_NODES * 4);
  int*    rp     = (int*)alloc((size_t)(N_NODES + 1) * 4);
  int*    pcol   = (int*)alloc((size_t)NE_MAX * 4);
  float*  pa     = (float*)alloc((size_t)NE_MAX * 4);
  float*  pw     = (float*)alloc((size_t)NE_MAX * 4);
  float*  f1     = (float*)alloc((size_t)N_NODES * NHID * 4);
  float*  f2     = (float*)alloc((size_t)N_NODES * NHID * 4);
  float*  f3     = (float*)alloc((size_t)N_NODES * NHID * 4);
  float*  f4     = (float*)alloc((size_t)N_NODES * NHID * 4);
  bf16_t* combbf = (bf16_t*)alloc((size_t)N_NODES * NHID * 2);

  // 1) precision conversion / weight transposes
  f32_to_bf16_kernel<<<(N_NODES * NFEAT + 255) / 256, 256, 0, stream>>>(x, x_bf, N_NODES * NFEAT);
  transpose_pad_bf16_kernel<<<(NHID * NFEAT + 255) / 256, 256, 0, stream>>>(W1, W1t_bf, NFEAT, NHID, NHID);
  transpose_pad_bf16_kernel<<<(48 * NHID + 255) / 256, 256, 0, stream>>>(W2, W2t_bf, NHID, NCLASS, 48);

  // 2) layer1: h = relu(x @ W1 + b1)   [WMMA bf16, 4 n-tiles per wave]
  gemm_bf16_wmma<4><<<dim3(NHID / 64, N_NODES / 128), 256, 0, stream>>>(
      x_bf, W1t_bf, b1, h, NFEAT, NHID, NHID, 1);

  // 3) two adj-SPMM hops: xa = A@h ; xb = A@xa
  hipMemsetAsync(xa, 0, (size_t)N_NODES * NHID * 4, stream);
  spmm_edge_kernel<<<(NNZ * 32 + 255) / 256, 256, 0, stream>>>(erow, ecolIn, avals, h, xa);
  hipMemsetAsync(xb, 0, (size_t)N_NODES * NHID * 4, stream);
  spmm_edge_kernel<<<(NNZ * 32 + 255) / 256, 256, 0, stream>>>(erow, ecolIn, avals, xa, xb);

  // 4) row-normalize -> xn (f32 + bf16)
  rownorm_kernel<<<N_NODES, NHID, 0, stream>>>(xb, xn, xn_bf);

  // 5) fused cos (WMMA) + per-row top-8, 32 rows/block (never materializes 256MB cos)
  cos_topk_kernel<<<N_NODES / 32, 256, 0, stream>>>(xn_bf, tkv, tki);

  // 6) edge cosine threshold values
  edge_val_kernel<<<(NNZ * 32 + 255) / 256, 256, 0, stream>>>(erow, ecolIn, avals, xn, eedge);

  // 7) build CSR of union pattern (edges + nonzero top-k)
  hipMemsetAsync(cnt, 0, (size_t)N_NODES * 4, stream);
  count_edges_kernel<<<(NNZ + 255) / 256, 256, 0, stream>>>(erow, cnt);
  count_topk_kernel<<<(N_NODES * KTOP + 255) / 256, 256, 0, stream>>>(tkv, cnt);
  scan_kernel<<<1, 1024, 0, stream>>>(cnt, rp, cur);
  fill_edges_kernel<<<(NNZ + 255) / 256, 256, 0, stream>>>(erow, ecolIn, eedge, cur, pcol, pa);
  fill_topk_kernel<<<(N_NODES * KTOP + 255) / 256, 256, 0, stream>>>(tkv, tki, cur, pcol, pa);

  // 8) sparse per-row softmax -> row-stochastic weights pw
  row_softmax_kernel<<<(N_NODES * 32 + 255) / 256, 256, 0, stream>>>(rp, pa, pw);

  // 9) 4 sparse propagation hops: f1..f4
  prop_kernel<<<N_NODES, NHID, 0, stream>>>(rp, pcol, pw, h, f1);
  prop_kernel<<<N_NODES, NHID, 0, stream>>>(rp, pcol, pw, f1, f2);
  prop_kernel<<<N_NODES, NHID, 0, stream>>>(rp, pcol, pw, f2, f3);
  prop_kernel<<<N_NODES, NHID, 0, stream>>>(rp, pcol, pw, f3, f4);

  // 10) per-node hop attention + combine
  hop_attn_kernel<<<N_NODES, NHID, 0, stream>>>(h, f1, f2, f3, f4, prp, combbf);

  // 11) layer2: out = comb @ W2 + b2   [WMMA bf16, N padded 40->48, 3 n-tiles per wave]
  gemm_bf16_wmma<3><<<dim3(1, N_NODES / 128), 256, 0, stream>>>(
      combbf, W2t_bf, b2, out, NHID, NCLASS, NCLASS, 0);
}
